// ColorGNN_47107201303213
// MI455X (gfx1250) — matrix operand order repaired and checked by reading, not compiled, not run on hardware
//
#include <hip/hip_runtime.h>
#include <stdint.h>

typedef __bf16 bf16;
typedef __attribute__((ext_vector_type(16))) bf16  v16bf;
typedef __attribute__((ext_vector_type(8)))  bf16  v8bf;
typedef __attribute__((ext_vector_type(8)))  float v8f;
typedef __attribute__((ext_vector_type(4)))  uint32_t v4u;
typedef __attribute__((ext_vector_type(4)))  int     i4v;

#define NBIRD 50000
#define PITCH 136    // bf16 tile row pitch (halves): 272B, 16B aligned, bank-conflict free
#define PITCHF 132   // f32 tile row pitch (floats): 528B, 16B aligned

#define AS1 __attribute__((address_space(1)))
#define AS3 __attribute__((address_space(3)))

#if defined(__has_builtin)
#if __has_builtin(__builtin_amdgcn_global_load_async_to_lds_b128)
#define HAVE_ASYNC_LDS 1
#endif
#endif

union F16 { v16bf v; v8bf h[2]; bf16 e[16]; };
union FC  { v8f  v; float e[8]; };

// 16B global -> LDS copy; async (ASYNCcnt) when the toolchain has the gfx1250 builtin.
__device__ __forceinline__ void cp16(void* lds, const void* glob) {
#ifdef HAVE_ASYNC_LDS
    __builtin_amdgcn_global_load_async_to_lds_b128(
        (AS1 i4v*)(uintptr_t)glob,
        (AS3 i4v*)(uint32_t)(uintptr_t)lds, 0, 0);
#else
    *(v4u*)lds = *(const v4u*)glob;
#endif
}
__device__ __forceinline__ void async_fence() {
#ifdef HAVE_ASYNC_LDS
#if __has_builtin(__builtin_amdgcn_s_wait_asynccnt)
    __builtin_amdgcn_s_wait_asynccnt(0);
#else
    asm volatile("s_wait_asynccnt 0x0" ::: "memory");
#endif
#endif
}

// A fragment (16xK tile, row-major buf): lane L holds row L%16; k = kb + (L>=16?8:0) + j (j<8)
// and k = kb + 16 + (L>=16?8:0) + (j-8) for j>=8  -> two b128 loads.
__device__ __forceinline__ v16bf fragA(const bf16* buf, int pitch, int row, int kb, int h16) {
    F16 f;
    const bf16* p = buf + row * pitch + kb + (h16 << 3);
    f.h[0] = *(const v8bf*)p;
    f.h[1] = *(const v8bf*)(p + 16);
    return f.v;
}
// B fragment from W^T stored [n][k]: lane L holds col n; k = kb + (L>=16?8:0) + j (j<8)
// and k = kb + 16 + (L>=16?8:0) + (j-8) for j>=8.
__device__ __forceinline__ v16bf fragB(const bf16* wt, int pitch, int n, int kb, int h16) {
    F16 f;
    const bf16* p = wt + n * pitch + kb + (h16 << 3);
    f.h[0] = *(const v8bf*)p;
    f.h[1] = *(const v8bf*)(p + 16);
    return f.v;
}
__device__ __forceinline__ v8f WMMA(v16bf a, v16bf b, v8f c) {
    return __builtin_amdgcn_wmma_f32_16x16x32_bf16(false, a, false, b, (short)0, c, false, false);
}
// intra-wave LDS RAW ordering (stage -> transpose-read by other lanes of same wave)
#define LDSFENCE() do { __builtin_amdgcn_wave_barrier(); \
    asm volatile("s_wait_dscnt 0x0" ::: "memory"); \
    __builtin_amdgcn_wave_barrier(); } while (0)

// ---------------- weight conversion: f32 row-major [K][N] -> bf16 transposed [N][K] -------------
__global__ __launch_bounds__(256) void k_weights(
    const float* eW1, const float* eW2, const float* nW1, const float* nW2,
    const float* node_W, const float* color_W,
    bf16* eW1T, bf16* eW2T, bf16* nW1T, bf16* nW2T, bf16* ndWT, bf16* clWT) {
    int i = blockIdx.x * 256 + threadIdx.x;
    const int S1 = 3 * 128 * 384, S2 = 3 * 128 * 128, S3 = 3 * 128 * 256, S4 = 3 * 128 * 128;
    const int S5 = 128 * 16, S6 = 16 * 128;
    if (i < S1) { int l = i / 49152, r = i % 49152, n = r / 384, k = r % 384;
        eW1T[i] = (bf16)eW1[(l * 384 + k) * 128 + n]; return; }
    i -= S1;
    if (i < S2) { int l = i / 16384, r = i % 16384, n = r / 128, k = r % 128;
        eW2T[i] = (bf16)eW2[(l * 128 + k) * 128 + n]; return; }
    i -= S2;
    if (i < S3) { int l = i / 32768, r = i % 32768, n = r / 256, k = r % 256;
        nW1T[i] = (bf16)nW1[(l * 256 + k) * 128 + n]; return; }
    i -= S3;
    if (i < S4) { int l = i / 16384, r = i % 16384, n = r / 128, k = r % 128;
        nW2T[i] = (bf16)nW2[(l * 128 + k) * 128 + n]; return; }
    i -= S4;
    if (i < S5) { int n = i / 16, k = i % 16; ndWT[i] = (bf16)node_W[k * 128 + n]; return; }
    i -= S5;
    if (i < S6) { int n = i / 128, k = i % 128; clWT[i] = (bf16)color_W[k * 16 + n]; return; }
}

// ------------- color init: Xc = node_W + node_b ; colorSum=0 ; colorTerm0 = Xc@W1b0 + eb1[0] ----
__global__ __launch_bounds__(256) void k_colorinit(
    const float* node_W, const float* node_b, const bf16* eW1T0, const float* eb1_0,
    bf16* XC, float* CS, float* CT) {
    __shared__ __align__(16) bf16 sXc[16 * PITCH];
    int t = threadIdx.x;
    for (int idx = t; idx < 2048; idx += 256) {
        int h = idx & 127;
        float v = node_W[idx] + node_b[h];
        XC[idx] = (bf16)v;
        sXc[(idx >> 7) * PITCH + h] = (bf16)v;
        CS[idx] = 0.f;
    }
    __syncthreads();
    int lane = t & 31, wv = t >> 5, l16 = lane & 15, h16 = lane >> 4;
    FC acc;
#pragma unroll
    for (int r = 0; r < 8; r++) acc.e[r] = eb1_0[wv * 16 + l16];
#pragma unroll
    for (int kt = 0; kt < 4; kt++) {
        v16bf a = fragA(sXc, PITCH, l16, kt * 32, h16);
        v16bf b = fragB(eW1T0, 384, wv * 16 + l16, 128 + kt * 32, h16);   // W1b block
        acc.v = WMMA(a, b, acc.v);
    }
#pragma unroll
    for (int r = 0; r < 8; r++) CT[(r + h16 * 8) * 128 + wv * 16 + l16] = acc.e[r];
}

// ------------- bird init: Xb = probs@node_W + node_b (WMMA, K padded) ; edge_attr rank-1 --------
__global__ __launch_bounds__(256) void k_birdinit(
    const float* probs, const float* node_b, const float* edge_W, const float* edge_b,
    const bf16* ndWT, bf16* XB, bf16* EA) {
    int t = threadIdx.x, lane = t & 31, wv = t >> 5, l16 = lane & 15, h16 = lane >> 4;
    int base = blockIdx.x * 16;
    F16 fa, fb;
    {
        const float* p = probs + (base + l16) * 16 + h16 * 8;
#pragma unroll
        for (int j = 0; j < 8; j++) fa.e[j] = (bf16)p[j];
#pragma unroll
        for (int j = 8; j < 16; j++) fa.e[j] = (bf16)0.0f;     // K=16 padded to 32
    }
    {
        const bf16* q = ndWT + (wv * 16 + l16) * 16 + h16 * 8;
        fb.h[0] = *(const v8bf*)q;
#pragma unroll
        for (int j = 8; j < 16; j++) fb.e[j] = (bf16)0.0f;     // k>=16 zero
    }
    FC acc;
#pragma unroll
    for (int r = 0; r < 8; r++) acc.e[r] = node_b[wv * 16 + l16];
    acc.v = WMMA(fa.v, fb.v, acc.v);
#pragma unroll
    for (int r = 0; r < 8; r++)
        XB[(base + r + h16 * 8) * 128 + wv * 16 + l16] = (bf16)acc.e[r];
    // edge_attr[i,c,:] = probs[i,c]*edge_W + edge_b
    for (int idx = t; idx < 16 * 2048; idx += 256) {
        int i = idx >> 11, c = (idx >> 7) & 15, h = idx & 127;
        float pv = probs[(base + i) * 16 + c];
        EA[(base + i) * 2048 + c * 128 + h] = (bf16)(pv * edge_W[h] + edge_b[h]);
    }
}

// ------------------------------ main fused edge+bird-node layer --------------------------------
__global__ __launch_bounds__(256) void k_edge(
    bf16* EA, bf16* XB, const float* CT, float* CS,
    const bf16* W1Tl, const bf16* W2Tl, const bf16* nW1Tl, const bf16* nW2Tl,
    const float* eb2l, const float* nb1l, const float* nb2l) {
    extern __shared__ char smem[];
    bf16*  sW1c = (bf16*)smem;             // 128 x PITCH  (W1c^T)
    bf16*  sW2  = sW1c + 128 * PITCH;      // 128 x PITCH  (eW2^T)
    bf16*  sXb  = sW2  + 128 * PITCH;      // 16 x PITCH   bird features (bf16)
    bf16*  sAgg = sXb  + 16 * PITCH;       // 16 x PITCH   per-bird edge aggregates
    bf16*  sH   = sAgg + 16 * PITCH;       // 8 waves x 16 x PITCH transpose staging
    float* sT   = (float*)(sH + 8 * 16 * PITCH); // 16 x PITCHF  T = Xb@W1a
    float* sCT  = sT  + 16 * PITCHF;       // 16 x PITCHF  colorTerm
    float* sCS  = sCT + 16 * PITCHF;       // 16 x PITCHF  color partial sums

    int t = threadIdx.x, lane = t & 31, wv = t >> 5, l16 = lane & 15, h16 = lane >> 4;
    int base = blockIdx.x * 16;

    // ---- stage weights/tiles (async global->LDS in 16B granules) ----
    for (int idx = t; idx < 2048; idx += 256) {          // W1c^T / eW2^T: 128 rows x 16 granules
        int n = idx >> 4, kc = (idx & 15) << 3;          // 8 halves per granule
        cp16(sW1c + n * PITCH + kc, W1Tl + n * 384 + 256 + kc);
        cp16(sW2  + n * PITCH + kc, W2Tl + n * 128 + kc);
    }
    {                                                    // Xb: 16 rows x 16 granules
        int m = t >> 4, kc = (t & 15) << 3;
        cp16(sXb + m * PITCH + kc, XB + (base + m) * 128 + kc);
    }
    for (int idx = t; idx < 512; idx += 256) {           // colorTerm f32: 16 rows x 32 granules
        int m = idx >> 5, kc = (idx & 31) << 2;          // 4 floats per granule
        cp16(sCT + m * PITCHF + kc, CT + m * 128 + kc);
    }
    for (int idx = t; idx < 16 * PITCHF; idx += 256) sCS[idx] = 0.f;
    async_fence();
    __syncthreads();

    // T = Xb @ W1a  (shared across the 16 birds of this block)
    {
        FC acc;
#pragma unroll
        for (int r = 0; r < 8; r++) acc.e[r] = 0.f;
#pragma unroll
        for (int kt = 0; kt < 4; kt++) {
            v16bf a = fragA(sXb, PITCH, l16, kt * 32, h16);
            v16bf b = fragB(W1Tl, 384, wv * 16 + l16, kt * 32, h16);  // W1a block
            acc.v = WMMA(a, b, acc.v);
        }
#pragma unroll
        for (int r = 0; r < 8; r++) sT[(r + h16 * 8) * PITCHF + wv * 16 + l16] = acc.e[r];
    }
    __syncthreads();

    float eb2v[8];
#pragma unroll
    for (int nt = 0; nt < 8; nt++) eb2v[nt] = eb2l[nt * 16 + l16];

    bf16* myH = sH + wv * 16 * PITCH;

    FC cacc[8];                                           // color partials, register-resident
#pragma unroll
    for (int nt = 0; nt < 8; nt++)
#pragma unroll
        for (int r = 0; r < 8; r++) cacc[nt].e[r] = 0.f;

    for (int round = 0; round < 2; ++round) {             // wave-per-bird, 2 birds per wave
        int i = wv + round * 8;
        int g = base + i;
        bf16* EAg = EA + g * 2048;
        if (round == 0)                                   // prefetch next bird's 4KB tile
            __builtin_prefetch(EAg + 8 * 2048 + lane * 64, 0, 3);
        FC acc[8];
#pragma unroll
        for (int nt = 0; nt < 8; nt++) {
            float tv = sT[i * PITCHF + nt * 16 + l16];
#pragma unroll
            for (int r = 0; r < 8; r++)
                acc[nt].e[r] = sCT[(r + h16 * 8) * PITCHF + nt * 16 + l16] + tv;
        }
#pragma unroll
        for (int kt = 0; kt < 4; kt++) {
            v16bf a = fragA(EAg, 128, l16, kt * 32, h16);       // edge_attr tile from HBM/L2
#pragma unroll
            for (int nt = 0; nt < 8; nt++) {
                v16bf b = fragB(sW1c, PITCH, nt * 16 + l16, kt * 32, h16);
                acc[nt].v = WMMA(a, b, acc[nt].v);
            }
        }
        // relu -> stage (C-layout -> row-major) for GEMM2 A operand
#pragma unroll
        for (int nt = 0; nt < 8; nt++)
#pragma unroll
            for (int r = 0; r < 8; r++)
                myH[(r + h16 * 8) * PITCH + nt * 16 + l16] = (bf16)fmaxf(acc[nt].e[r], 0.f);
        LDSFENCE();
        // ea' = h @ eW2 + eb2
#pragma unroll
        for (int nt = 0; nt < 8; nt++)
#pragma unroll
            for (int r = 0; r < 8; r++) acc[nt].e[r] = eb2v[nt];
#pragma unroll
        for (int kt = 0; kt < 4; kt++) {
            v16bf a = fragA(myH, PITCH, l16, kt * 32, h16);
#pragma unroll
            for (int nt = 0; nt < 8; nt++) {
                v16bf b = fragB(sW2, PITCH, nt * 16 + l16, kt * 32, h16);
                acc[nt].v = WMMA(a, b, acc[nt].v);
            }
        }
        // color partials into registers + bird aggregate (column sums via shfl)
#pragma unroll
        for (int nt = 0; nt < 8; nt++) {
            float s = 0.f;
#pragma unroll
            for (int r = 0; r < 8; r++) {
                float v = acc[nt].e[r];
                s += v;
                cacc[nt].e[r] += v;
            }
            s += __shfl_xor(s, 16);
            if (!h16) sAgg[i * PITCH + nt * 16 + l16] = (bf16)s;
        }
        // stage ea' and store back coalesced (b128)
#pragma unroll
        for (int nt = 0; nt < 8; nt++)
#pragma unroll
            for (int r = 0; r < 8; r++)
                myH[(r + h16 * 8) * PITCH + nt * 16 + l16] = (bf16)acc[nt].e[r];
        LDSFENCE();
#pragma unroll
        for (int p = 0; p < 8; p++) {
            int m = (p << 1) | h16;
            *(v8bf*)(EAg + m * 128 + l16 * 8) = *(const v8bf*)(myH + m * PITCH + l16 * 8);
        }
    }
    // one LDS-atomic pass for both birds' color partials
#pragma unroll
    for (int nt = 0; nt < 8; nt++)
#pragma unroll
        for (int r = 0; r < 8; r++)
            atomicAdd(&sCS[(r + h16 * 8) * PITCHF + nt * 16 + l16], cacc[nt].e[r]);
    __syncthreads();

    // bird node MLP: x' = relu([Xb|Agg] @ nW1 + nb1) @ nW2 + nb2   (concat = k-step selection)
    {
        FC acc;
#pragma unroll
        for (int r = 0; r < 8; r++) acc.e[r] = nb1l[wv * 16 + l16];
#pragma unroll
        for (int kt = 0; kt < 8; kt++) {
            v16bf a = (kt < 4) ? fragA(sXb, PITCH, l16, kt * 32, h16)
                               : fragA(sAgg, PITCH, l16, (kt - 4) * 32, h16);
            v16bf b = fragB(nW1Tl, 256, wv * 16 + l16, kt * 32, h16);
            acc.v = WMMA(a, b, acc.v);
        }
        bf16* sHn = (bf16*)sT;   // reuse: T no longer needed
#pragma unroll
        for (int r = 0; r < 8; r++)
            sHn[(r + h16 * 8) * PITCH + wv * 16 + l16] = (bf16)fmaxf(acc.e[r], 0.f);
        __syncthreads();
        FC a2;
#pragma unroll
        for (int r = 0; r < 8; r++) a2.e[r] = nb2l[wv * 16 + l16];
#pragma unroll
        for (int kt = 0; kt < 4; kt++) {
            v16bf a = fragA(sHn, PITCH, l16, kt * 32, h16);
            v16bf b = fragB(nW2Tl, 128, wv * 16 + l16, kt * 32, h16);
            a2.v = WMMA(a, b, a2.v);
        }
#pragma unroll
        for (int r = 0; r < 8; r++)
            XB[(base + r + h16 * 8) * 128 + wv * 16 + l16] = (bf16)a2.e[r];
    }
    // flush color partials to global
    for (int idx = t; idx < 2048; idx += 256)
        atomicAdd(&CS[idx], sCS[(idx >> 7) * PITCHF + (idx & 127)]);
}

// ------------- color node update + next-layer colorTerm (single block) -------------------------
__global__ __launch_bounds__(256) void k_color(
    bf16* XC, float* CS, float* CT,
    const bf16* nW1Tl, const bf16* nW2Tl, const float* nb1l, const float* nb2l,
    const bf16* eW1Tn, const float* eb1n, int doNext) {
    __shared__ __align__(16) bf16 sXc[16 * PITCH], sAg[16 * PITCH], sHn[16 * PITCH], sXn[16 * PITCH];
    int t = threadIdx.x, lane = t & 31, wv = t >> 5, l16 = lane & 15, h16 = lane >> 4;
    for (int idx = t; idx < 2048; idx += 256) {
        int c = idx >> 7, h = idx & 127;
        sXc[c * PITCH + h] = XC[idx];
        sAg[c * PITCH + h] = (bf16)CS[idx];
        CS[idx] = 0.f;                      // re-arm for next layer's atomics
    }
    __syncthreads();
    FC acc;
#pragma unroll
    for (int r = 0; r < 8; r++) acc.e[r] = nb1l[wv * 16 + l16];
#pragma unroll
    for (int kt = 0; kt < 8; kt++) {
        v16bf a = (kt < 4) ? fragA(sXc, PITCH, l16, kt * 32, h16)
                           : fragA(sAg, PITCH, l16, (kt - 4) * 32, h16);
        v16bf b = fragB(nW1Tl, 256, wv * 16 + l16, kt * 32, h16);
        acc.v = WMMA(a, b, acc.v);
    }
#pragma unroll
    for (int r = 0; r < 8; r++)
        sHn[(r + h16 * 8) * PITCH + wv * 16 + l16] = (bf16)fmaxf(acc.e[r], 0.f);
    __syncthreads();
    FC a2;
#pragma unroll
    for (int r = 0; r < 8; r++) a2.e[r] = nb2l[wv * 16 + l16];
#pragma unroll
    for (int kt = 0; kt < 4; kt++) {
        v16bf a = fragA(sHn, PITCH, l16, kt * 32, h16);
        v16bf b = fragB(nW2Tl, 128, wv * 16 + l16, kt * 32, h16);
        a2.v = WMMA(a, b, a2.v);
    }
#pragma unroll
    for (int r = 0; r < 8; r++) {
        int m = r + h16 * 8;
        float v = a2.e[r];
        XC[m * 128 + wv * 16 + l16] = (bf16)v;
        sXn[m * PITCH + wv * 16 + l16] = (bf16)v;
    }
    __syncthreads();
    if (doNext) {
        FC a3;
#pragma unroll
        for (int r = 0; r < 8; r++) a3.e[r] = eb1n[wv * 16 + l16];
#pragma unroll
        for (int kt = 0; kt < 4; kt++) {
            v16bf a = fragA(sXn, PITCH, l16, kt * 32, h16);
            v16bf b = fragB(eW1Tn, 384, wv * 16 + l16, 128 + kt * 32, h16);  // W1b block
            a3.v = WMMA(a, b, a3.v);
        }
#pragma unroll
        for (int r = 0; r < 8; r++) CT[(r + h16 * 8) * 128 + wv * 16 + l16] = a3.e[r];
    }
}

// ------------- final: scores = Xb @ color_W + color_b ; out = scores * probs -------------------
__global__ __launch_bounds__(256) void k_final(
    const bf16* XB, const bf16* clWT, const float* color_b, const float* probs, float* out) {
    int t = threadIdx.x, lane = t & 31, wv = t >> 5, l16 = lane & 15, h16 = lane >> 4;
    int tile = blockIdx.x * 8 + wv;
    if (tile >= 3125) return;               // wave-uniform
    int base = tile * 16;
    FC acc;
#pragma unroll
    for (int r = 0; r < 8; r++) acc.e[r] = color_b[l16];
#pragma unroll
    for (int kt = 0; kt < 4; kt++) {
        v16bf a = fragA(XB + base * 128, 128, l16, kt * 32, h16);
        v16bf b = fragB(clWT, 128, l16, kt * 32, h16);
        acc.v = WMMA(a, b, acc.v);
    }
#pragma unroll
    for (int r = 0; r < 8; r++) {
        int gi = (base + r + h16 * 8) * 16 + l16;
        out[gi] = acc.e[r] * probs[gi];
    }
}

extern "C" void kernel_launch(void* const* d_in, const int* in_sizes, int n_in,
                              void* d_out, int out_size, void* d_ws, size_t ws_size,
                              hipStream_t stream) {
    const float* probs   = (const float*)d_in[0];
    const float* node_W  = (const float*)d_in[1];
    const float* node_b  = (const float*)d_in[2];
    const float* edge_W  = (const float*)d_in[3];
    const float* edge_b  = (const float*)d_in[4];
    const float* eW1     = (const float*)d_in[5];
    const float* eb1     = (const float*)d_in[6];
    const float* eW2     = (const float*)d_in[7];
    const float* eb2     = (const float*)d_in[8];
    const float* nW1     = (const float*)d_in[9];
    const float* nb1     = (const float*)d_in[10];
    const float* nW2     = (const float*)d_in[11];
    const float* nb2     = (const float*)d_in[12];
    const float* color_W = (const float*)d_in[13];
    const float* color_b = (const float*)d_in[14];

    bf16*  EA   = (bf16*)d_ws;                       // [N][16][128] bf16 edge_attr (205 MB)
    bf16*  XB   = EA + (size_t)NBIRD * 2048;         // [N][128] bird features
    bf16*  XC   = XB + (size_t)NBIRD * 128;          // [16][128] color features
    float* CS   = (float*)(XC + 2048);               // [16][128] color scatter-sum
    float* CT   = CS + 2048;                         // [16][128] colorTerm
    bf16*  eW1T = (bf16*)(CT + 2048);                // bf16 transposed weights
    bf16*  eW2T = eW1T + (size_t)3 * 128 * 384;
    bf16*  nW1T = eW2T + (size_t)3 * 128 * 128;
    bf16*  nW2T = nW1T + (size_t)3 * 128 * 256;
    bf16*  ndWT = nW2T + (size_t)3 * 128 * 128;
    bf16*  clWT = ndWT + 128 * 16;

    const size_t SHEDGE =
        (size_t)(2 * 128 * PITCH + 2 * 16 * PITCH + 8 * 16 * PITCH) * sizeof(bf16) +
        (size_t)3 * 16 * PITCHF * sizeof(float);     // 138,496 B dynamic LDS

    k_weights<<<1360, 256, 0, stream>>>(eW1, eW2, nW1, nW2, node_W, color_W,
                                        eW1T, eW2T, nW1T, nW2T, ndWT, clWT);
    k_colorinit<<<1, 256, 0, stream>>>(node_W, node_b, eW1T, eb1, XC, CS, CT);
    k_birdinit<<<3125, 256, 0, stream>>>(probs, node_b, edge_W, edge_b, ndWT, XB, EA);
    for (int l = 0; l < 3; ++l) {
        k_edge<<<3125, 256, SHEDGE, stream>>>(EA, XB, CT, CS,
            eW1T + (size_t)l * 128 * 384, eW2T + (size_t)l * 128 * 128,
            nW1T + (size_t)l * 128 * 256, nW2T + (size_t)l * 128 * 128,
            eb2 + l * 128, nb1 + l * 128, nb2 + l * 128);
        if (l < 2)   // last color-node update is dead code w.r.t. the output
            k_color<<<1, 256, 0, stream>>>(XC, CS, CT,
                nW1T + (size_t)l * 128 * 256, nW2T + (size_t)l * 128 * 128,
                nb1 + l * 128, nb2 + l * 128,
                eW1T + (size_t)(l + 1) * 128 * 384, eb1 + (l + 1) * 128, 1);
    }
    k_final<<<391, 256, 0, stream>>>(XB, clWT, color_b, probs, (float*)d_out);
}
